// SingleEncoding__32993938767929
// MI455X (gfx1250) — matrix operand order repaired and checked by reading, not compiled, not run on hardware
//
#include <hip/hip_runtime.h>
#include <hip/hip_bf16.h>

// ---------------- types for WMMA ----------------
typedef __attribute__((ext_vector_type(16))) _Float16 v16h;
typedef __attribute__((ext_vector_type(8)))  _Float16 v8h;
typedef __attribute__((ext_vector_type(8)))  float    v8f;

// ---------------- problem constants ----------------
#define BATCH   2048
#define IMGD    28
#define HP      14              // patches per side
#define NPATCH  (HP * HP)       // 196 patches per image
#define FEAT    784             // 196 * 4
#define KPAD    800             // FEAT padded to multiple of 32
#define NHID    20
#define NHIDPAD 32
#define NCLS    2

// ws layout (bytes)
#define ZH_BYTES  ((size_t)BATCH * KPAD * 2)         // 3,276,800
#define WH_BYTES  ((size_t)NHIDPAD * KPAD * 2)       // 51,200

// =====================================================================
// Kernel 1: convert fc1_w (20x784 f32) -> f16 padded [32][800], and
//           zero the K-pad columns (784..799) of Zh.
// =====================================================================
__global__ void prep_kernel(const float* __restrict__ fc1_w,
                            _Float16* __restrict__ Wh,
                            _Float16* __restrict__ Zh) {
    int t = blockIdx.x * blockDim.x + threadIdx.x;
    const int NW   = NHIDPAD * KPAD;    // 25600
    const int NPADZ = BATCH * (KPAD - FEAT); // 2048*16 = 32768
    if (t < NW) {
        int n = t / KPAD, k = t % KPAD;
        float v = (n < NHID && k < FEAT) ? fc1_w[n * FEAT + k] : 0.0f;
        Wh[n * KPAD + k] = (_Float16)v;
    } else if (t < NW + NPADZ) {
        int u = t - NW;
        int r = u >> 4, c = u & 15;
        Zh[r * KPAD + FEAT + c] = (_Float16)0.0f;
    }
}

// =====================================================================
// Kernel 2: 4-qubit circuit per 2x2 patch. State (16 complex amps)
// lives entirely in VGPRs; fully unrolled so all masks constant-fold.
// Wire w maps to state-index bit (3-w):  idx = b0*8+b1*4+b2*2+b3.
// =====================================================================
__global__ void circuit_kernel(const float* __restrict__ x,
                               const float* __restrict__ ry_theta,
                               const float* __restrict__ crz_theta,
                               _Float16* __restrict__ Zh) {
    int p = blockIdx.x * blockDim.x + threadIdx.x;
    if (p >= BATCH * NPATCH) return;
    int b  = p / NPATCH;
    int pi = p - b * NPATCH;
    int i  = pi / HP;
    int j  = pi - i * HP;

    const float* xb = x + (size_t)b * (IMGD * IMGD);
    // patch pixel order (di,dj) = (0,0),(0,1),(1,0),(1,1) -> wires 0..3
    float a0 = xb[(2 * i    ) * IMGD + 2 * j    ];
    float a1 = xb[(2 * i    ) * IMGD + 2 * j + 1];
    float a2 = xb[(2 * i + 1) * IMGD + 2 * j    ];
    float a3 = xb[(2 * i + 1) * IMGD + 2 * j + 1];

    float ca[4], sa[4];
    ca[0] = cosf(0.5f * a0); sa[0] = sinf(0.5f * a0);
    ca[1] = cosf(0.5f * a1); sa[1] = sinf(0.5f * a1);
    ca[2] = cosf(0.5f * a2); sa[2] = sinf(0.5f * a2);
    ca[3] = cosf(0.5f * a3); sa[3] = sinf(0.5f * a3);

    float re[16], im[16];
#pragma unroll
    for (int s = 0; s < 16; ++s) {
        float v = ((s & 8) ? sa[0] : ca[0]) *
                  ((s & 4) ? sa[1] : ca[1]) *
                  ((s & 2) ? sa[2] : ca[2]) *
                  ((s & 1) ? sa[3] : ca[3]);
        re[s] = v;
        im[s] = 0.0f;
    }

    float thy = 0.5f * ry_theta[0];
    float thc = 0.5f * crz_theta[0];
    float cy = cosf(thy), sy = sinf(thy);
    float cr = cosf(thc), sr = sinf(thc);

#pragma unroll
    for (int d = 0; d < 2; ++d) {
        // ring of CRZ gates: control g, target (g+1)%4
#pragma unroll
        for (int g = 0; g < 4; ++g) {
            const int cm = 8 >> g;
            const int tm = 8 >> ((g + 1) & 3);
#pragma unroll
            for (int s = 0; s < 16; ++s) {
                if (s & cm) {
                    float sp = (s & tm) ? sr : -sr;   // exp(i*th/2*(2bt-1))
                    float r = re[s], q = im[s];
                    re[s] = r * cr - q * sp;
                    im[s] = r * sp + q * cr;
                }
            }
        }
        // RY on every wire
#pragma unroll
        for (int w = 0; w < 4; ++w) {
            const int m = 8 >> w;
#pragma unroll
            for (int s = 0; s < 16; ++s) {
                if (!(s & m)) {
                    const int s1 = s | m;
                    float r0 = re[s], r1 = re[s1];
                    float i0 = im[s], i1 = im[s1];
                    re[s]  = cy * r0 - sy * r1;
                    re[s1] = sy * r0 + cy * r1;
                    im[s]  = cy * i0 - sy * i1;
                    im[s1] = sy * i0 + cy * i1;
                }
            }
        }
    }

    float z0 = 0.f, z1 = 0.f, z2 = 0.f, z3 = 0.f;
#pragma unroll
    for (int s = 0; s < 16; ++s) {
        float pr = re[s] * re[s] + im[s] * im[s];
        z0 += (s & 8) ? -pr : pr;
        z1 += (s & 4) ? -pr : pr;
        z2 += (s & 2) ? -pr : pr;
        z3 += (s & 1) ? -pr : pr;
    }

    _Float16* dst = Zh + (size_t)b * KPAD + pi * 4;
    dst[0] = (_Float16)z0;
    dst[1] = (_Float16)z1;
    dst[2] = (_Float16)z2;
    dst[3] = (_Float16)z3;
}

// =====================================================================
// Kernel 3: out = relu(Z @ W1^T + b1) @ W2^T + b2 via v_wmma_f32_16x16x32_f16
// One wave (32 threads) per 16-row batch tile; two N-tiles (cols 0..15,
// 16..31; weights zero-padded past 20).
// =====================================================================
__global__ void fc_wmma_kernel(const _Float16* __restrict__ Zh,
                               const _Float16* __restrict__ Wh,
                               const float* __restrict__ fc1_b,
                               const float* __restrict__ fc2_w,
                               const float* __restrict__ fc2_b,
                               float* __restrict__ out) {
    __shared__ float hid[16][NHID];
    const int lane = threadIdx.x;          // 0..31
    const int row0 = blockIdx.x * 16;      // batch tile start
    const int nb   = lane & 15;
    const int half = lane >> 4;

    v8f c0 = {};   // hidden cols 0..15
    v8f c1 = {};   // hidden cols 16..31 (only 16..19 real)

    const _Float16* arow = Zh + (size_t)(row0 + nb) * KPAD;  // M = row0 + (lane&15)
    const _Float16* b0row = Wh + (size_t)nb * KPAD;          // N = lane&15
    const _Float16* b1row = Wh + (size_t)(16 + nb) * KPAD;   // N = 16 + (lane&15)

    for (int kb = 0; kb < KPAD; kb += 32) {
        // A fragment (16x32 f16): lane-half 0 holds K 0-15; half 1 holds 16-31 (contiguous)
        union { v16h v; v8h h[2]; } a;
        a.h[0] = *(const v8h*)(arow + kb + half * 16);
        a.h[1] = *(const v8h*)(arow + kb + half * 16 + 8);
        // B fragment (32x16 f16): lane = column, half selects K 0-15 vs 16-31 (contiguous)
        v16h b0 = *(const v16h*)(b0row + kb + half * 16);
        v16h b1 = *(const v16h*)(b1row + kb + half * 16);

        c0 = __builtin_amdgcn_wmma_f32_16x16x32_f16(false, a.v, false, b0,
                                                    (short)0, c0, false, false);
        c1 = __builtin_amdgcn_wmma_f32_16x16x32_f16(false, a.v, false, b1,
                                                    (short)0, c1, false, false);
    }

    // C/D layout: VGPR r holds M = r + 8*half, N = lane&15
    float bias0 = fc1_b[nb];
    float bias1 = (nb < 4) ? fc1_b[16 + nb] : 0.0f;
#pragma unroll
    for (int r = 0; r < 8; ++r) {
        int mr = r + 8 * half;
        float h0 = c0[r] + bias0;
        hid[mr][nb] = h0 > 0.f ? h0 : 0.f;
        if (nb < 4) {
            float h1 = c1[r] + bias1;
            hid[mr][16 + nb] = h1 > 0.f ? h1 : 0.f;
        }
    }
    __syncthreads();

    if (lane < 16) {
        float acc0 = fc2_b[0];
        float acc1 = fc2_b[1];
#pragma unroll
        for (int n = 0; n < NHID; ++n) {
            float h = hid[lane][n];
            acc0 += h * fc2_w[n];
            acc1 += h * fc2_w[NHID + n];
        }
        out[(size_t)(row0 + lane) * NCLS + 0] = acc0;
        out[(size_t)(row0 + lane) * NCLS + 1] = acc1;
    }
}

// =====================================================================
extern "C" void kernel_launch(void* const* d_in, const int* in_sizes, int n_in,
                              void* d_out, int out_size, void* d_ws, size_t ws_size,
                              hipStream_t stream) {
    const float* x      = (const float*)d_in[0];
    const float* ry_th  = (const float*)d_in[1];
    const float* crz_th = (const float*)d_in[2];
    const float* fc1_w  = (const float*)d_in[3];
    const float* fc1_b  = (const float*)d_in[4];
    const float* fc2_w  = (const float*)d_in[5];
    const float* fc2_b  = (const float*)d_in[6];
    float* out = (float*)d_out;

    _Float16* Zh = (_Float16*)d_ws;
    _Float16* Wh = (_Float16*)((char*)d_ws + ZH_BYTES);

    // 1) weight convert + Zh K-pad zeroing
    {
        int total = NHIDPAD * KPAD + BATCH * (KPAD - FEAT);  // 58368
        int blk = 256;
        prep_kernel<<<(total + blk - 1) / blk, blk, 0, stream>>>(fc1_w, Wh, Zh);
    }
    // 2) per-patch 4-qubit circuit -> f16 features
    {
        int total = BATCH * NPATCH;                          // 401408
        int blk = 256;
        circuit_kernel<<<(total + blk - 1) / blk, blk, 0, stream>>>(x, ry_th, crz_th, Zh);
    }
    // 3) WMMA FC1 + relu + FC2
    {
        fc_wmma_kernel<<<BATCH / 16, 32, 0, stream>>>(Zh, Wh, fc1_b, fc2_w, fc2_b, out);
    }
}
